// GCN_38448547234466
// MI455X (gfx1250) — compile-verified, matrix operand order checked by reading, not run on hardware
//
#include <hip/hip_runtime.h>
#include <stdint.h>
#include <stddef.h>

// ---------------------------------------------------------------------------
// GCN (2-layer, PyG GCNConv semantics) for MI455X / gfx1250, wave32 + WMMA.
// ---------------------------------------------------------------------------

#define N_NODES   50000
#define N_EDGES   1600000
#define F_IN      1433
#define HID       256
#define C_OUT     7
#define NPAD2     64      // padded N for second GEMM

typedef __attribute__((ext_vector_type(16))) __bf16 v16bf;
typedef __attribute__((ext_vector_type(8)))  __bf16 v8bf;
typedef __attribute__((ext_vector_type(8)))  float  v8f;

__device__ __forceinline__ unsigned short f2bf(float f) {
  union { float f; unsigned int u; } c; c.f = f;
  unsigned int u = c.u;
  unsigned int r = u + 0x7FFFu + ((u >> 16) & 1u);   // round-to-nearest-even
  return (unsigned short)(r >> 16);
}
__device__ __forceinline__ float bf2f(unsigned short h) {
  union { unsigned int u; float f; } c; c.u = ((unsigned int)h) << 16;
  return c.f;
}

// ---------------------------------------------------------------------------
// Split-bf16 WMMA GEMM:  C[M,N] = A[M,K] (lda) * B[K,N] (ldb), fp32 in/out.
// x = hi + lo (both bf16); product = hi*hi + hi*lo + lo*hi, fp32 accumulate.
// BM=128, BN=64, BK=32, 256 threads (8 waves); wave w owns rows [16w,16w+16)
// and all four 16-wide N subtiles.
// ---------------------------------------------------------------------------
#define BM 128
#define BN 64
#define BK 32
#define LDSA 40   // padded row stride in ushorts -> conflict-free b128 reads
#define LDSB 40

__global__ __launch_bounds__(256)
void gcn_gemm_bf16x3(const float* __restrict__ A, int lda,
                     const float* __restrict__ B, int ldb,
                     float* __restrict__ C, int ldc,
                     int M, int N, int K)
{
  __shared__ unsigned short sAhi[BM * LDSA];
  __shared__ unsigned short sAlo[BM * LDSA];
  __shared__ unsigned short sBhi[BN * LDSB];
  __shared__ unsigned short sBlo[BN * LDSB];

  const int t    = threadIdx.x;
  const int lane = t & 31;
  const int wave = t >> 5;
  const int M0   = blockIdx.x * BM;
  const int N0   = blockIdx.y * BN;

  const int mrow = lane & 15;   // N index in B/C fragments, M row in A frag
  const int half = lane >> 4;   // K-half selector per ISA 16-bit layouts

  const bool fullM = (M0 + BM) <= M;

  v8f acc[4];
  #pragma unroll
  for (int j = 0; j < 4; ++j) acc[j] = (v8f){0.f,0.f,0.f,0.f,0.f,0.f,0.f,0.f};

  const int kTiles = (K + BK - 1) / BK;
  for (int kt = 0; kt < kTiles; ++kt) {
    const int k0 = kt * BK;
    const bool fullK = (k0 + BK) <= K;

    if (fullM && fullK) {
      // ---------------- fast path: whole tile in bounds, no guards ----------
      #pragma unroll
      for (int i = 0; i < 16; ++i) {
        int idx = t + i * 256;          // 4096 elements: BM x BK
        int row = idx >> 5;
        int kk  = idx & 31;
        float x = A[(size_t)(M0 + row) * lda + (k0 + kk)];
        unsigned short hi = f2bf(x);
        unsigned short lo = f2bf(x - bf2f(hi));
        sAhi[row * LDSA + kk] = hi;
        sAlo[row * LDSA + kk] = lo;
      }
      #pragma unroll
      for (int i = 0; i < 8; ++i) {
        int idx = t + i * 256;          // 2048 elements: BK x BN
        int kk  = idx >> 6;
        int n   = idx & 63;
        float x = B[(size_t)(k0 + kk) * ldb + (N0 + n)];
        unsigned short hi = f2bf(x);
        unsigned short lo = f2bf(x - bf2f(hi));
        sBhi[n * LDSB + kk] = hi;
        sBlo[n * LDSB + kk] = lo;
      }
    } else {
      // ------------- edge path: clamp address, select zero (no branches) ----
      #pragma unroll
      for (int i = 0; i < 16; ++i) {
        int idx = t + i * 256;
        int row = idx >> 5;
        int kk  = idx & 31;
        int gr  = M0 + row;
        int gk  = k0 + kk;
        int grc = gr < (M - 1) ? gr : (M - 1);
        int gkc = gk < (K - 1) ? gk : (K - 1);
        float x = A[(size_t)grc * lda + gkc];
        if (gr >= M || gk >= K) x = 0.0f;
        unsigned short hi = f2bf(x);
        unsigned short lo = f2bf(x - bf2f(hi));
        sAhi[row * LDSA + kk] = hi;
        sAlo[row * LDSA + kk] = lo;
      }
      #pragma unroll
      for (int i = 0; i < 8; ++i) {
        int idx = t + i * 256;
        int kk  = idx >> 6;
        int n   = idx & 63;
        int gk  = k0 + kk;
        int gkc = gk < (K - 1) ? gk : (K - 1);
        float x = B[(size_t)gkc * ldb + (N0 + n)];
        if (gk >= K) x = 0.0f;
        unsigned short hi = f2bf(x);
        unsigned short lo = f2bf(x - bf2f(hi));
        sBhi[n * LDSB + kk] = hi;
        sBlo[n * LDSB + kk] = lo;
      }
    }
    __syncthreads();

    // ---- A fragments: lane holds row (16w + mrow); K runs 8h..8h+7, 16+8h.. ----
    const int arow = wave * 16 + mrow;
    const unsigned short* pah = &sAhi[arow * LDSA + 8 * half];
    const unsigned short* pal = &sAlo[arow * LDSA + 8 * half];
    v8bf ah0 = *reinterpret_cast<const v8bf*>(pah);
    v8bf ah1 = *reinterpret_cast<const v8bf*>(pah + 16);
    v8bf al0 = *reinterpret_cast<const v8bf*>(pal);
    v8bf al1 = *reinterpret_cast<const v8bf*>(pal + 16);
    v16bf ahi = __builtin_shufflevector(ah0, ah1, 0,1,2,3,4,5,6,7,8,9,10,11,12,13,14,15);
    v16bf alo = __builtin_shufflevector(al0, al1, 0,1,2,3,4,5,6,7,8,9,10,11,12,13,14,15);

    #pragma unroll
    for (int j = 0; j < 4; ++j) {
      const unsigned short* pbh = &sBhi[(j * 16 + mrow) * LDSB + 8 * half];
      const unsigned short* pbl = &sBlo[(j * 16 + mrow) * LDSB + 8 * half];
      v8bf bh0 = *reinterpret_cast<const v8bf*>(pbh);
      v8bf bh1 = *reinterpret_cast<const v8bf*>(pbh + 16);
      v8bf bl0 = *reinterpret_cast<const v8bf*>(pbl);
      v8bf bl1 = *reinterpret_cast<const v8bf*>(pbl + 16);
      v16bf bhi = __builtin_shufflevector(bh0, bh1, 0,1,2,3,4,5,6,7,8,9,10,11,12,13,14,15);
      v16bf blo = __builtin_shufflevector(bl0, bl1, 0,1,2,3,4,5,6,7,8,9,10,11,12,13,14,15);

      acc[j] = __builtin_amdgcn_wmma_f32_16x16x32_bf16(false, ahi, false, bhi,
                                                       (short)0, acc[j], false, false);
      acc[j] = __builtin_amdgcn_wmma_f32_16x16x32_bf16(false, ahi, false, blo,
                                                       (short)0, acc[j], false, false);
      acc[j] = __builtin_amdgcn_wmma_f32_16x16x32_bf16(false, alo, false, bhi,
                                                       (short)0, acc[j], false, false);
    }
    __syncthreads();
  }

  // ---- epilogue: D layout — VGPR v, lane L: M' = v + 8*(L>=16), N' = L&15 ----
  #pragma unroll
  for (int j = 0; j < 4; ++j) {
    #pragma unroll
    for (int v = 0; v < 8; ++v) {
      int gr = M0 + wave * 16 + v + 8 * half;
      int gc = N0 + j * 16 + mrow;
      if (gr < M) C[(size_t)gr * ldc + gc] = acc[j][v];
    }
  }
}

// ---------------------------------------------------------------------------
// Degree / normalization kernels
// ---------------------------------------------------------------------------
__global__ void k_init_deg(float* deg, int n) {
  int i = blockIdx.x * blockDim.x + threadIdx.x;
  if (i < n) deg[i] = 1.0f;                 // self-loop
}
__global__ void k_deg_edges(const int* __restrict__ dst, float* deg, int E) {
  int e = blockIdx.x * blockDim.x + threadIdx.x;
  if (e < E) atomicAdd(&deg[dst[e]], 1.0f);
}
__global__ void k_dinv(const float* __restrict__ deg, float* dinv, int n) {
  int i = blockIdx.x * blockDim.x + threadIdx.x;
  if (i < n) dinv[i] = __frsqrt_rn(deg[i]); // deg >= 1 always
}

// W2 [256,7] -> padded [256,64]
__global__ void k_pad_w2(const float* __restrict__ W2, float* __restrict__ W2p) {
  int idx = blockIdx.x * blockDim.x + threadIdx.x;
  if (idx < HID * NPAD2) {
    int k = idx / NPAD2, n = idx % NPAD2;
    W2p[idx] = (n < C_OUT) ? W2[k * C_OUT + n] : 0.0f;
  }
}

// ---------------------------------------------------------------------------
// Aggregation (scatter-add with symmetric norm), layer 1: F = 256
// Both h1 (51 MB) and agg1 (51 MB) fit in the 192 MB L2, so the float
// atomics resolve in L2 — plain atomic scatter is the right structure here.
// ---------------------------------------------------------------------------
__global__ __launch_bounds__(256)
void k_selfloop1(const float* __restrict__ h, const float* __restrict__ dinv,
                 float* __restrict__ agg) {
  int i = blockIdx.x;              // node
  int f = threadIdx.x;             // feature
  float w = dinv[i] * dinv[i];
  agg[(size_t)i * HID + f] = h[(size_t)i * HID + f] * w;
}

#define EDGES_PER_BLOCK 16
__global__ __launch_bounds__(256)
void k_scatter1(const int* __restrict__ src, const int* __restrict__ dst,
                const float* __restrict__ dinv, const float* __restrict__ h,
                float* __restrict__ agg, int E) {
  int f = threadIdx.x;
  int base = blockIdx.x * EDGES_PER_BLOCK;
  #pragma unroll
  for (int t = 0; t < EDGES_PER_BLOCK; ++t) {
    int e = base + t;
    if (e < E) {
      int s = src[e], d = dst[e];
      float w = dinv[s] * dinv[d];
      atomicAdd(&agg[(size_t)d * HID + f], h[(size_t)s * HID + f] * w);
    }
  }
}

__global__ __launch_bounds__(256)
void k_bias_relu1(const float* __restrict__ agg, const float* __restrict__ b,
                  float* __restrict__ out) {
  int i = blockIdx.x;
  int f = threadIdx.x;
  out[(size_t)i * HID + f] = fmaxf(agg[(size_t)i * HID + f] + b[f], 0.0f);
}

// ---------------------------------------------------------------------------
// Aggregation, layer 2: F = 7 (g stored with row stride NPAD2)
// ---------------------------------------------------------------------------
__global__ void k_selfloop2(const float* __restrict__ g, const float* __restrict__ dinv,
                            float* __restrict__ agg2) {
  int idx = blockIdx.x * blockDim.x + threadIdx.x;
  if (idx < N_NODES * C_OUT) {
    int i = idx / C_OUT, c = idx - i * C_OUT;
    float w = dinv[i] * dinv[i];
    agg2[idx] = g[(size_t)i * NPAD2 + c] * w;
  }
}

__global__ void k_scatter2(const int* __restrict__ src, const int* __restrict__ dst,
                           const float* __restrict__ dinv, const float* __restrict__ g,
                           float* __restrict__ agg2, int E) {
  int e = blockIdx.x * blockDim.x + threadIdx.x;
  if (e < E) {
    int s = src[e], d = dst[e];
    float w = dinv[s] * dinv[d];
    #pragma unroll
    for (int c = 0; c < C_OUT; ++c)
      atomicAdd(&agg2[(size_t)d * C_OUT + c], g[(size_t)s * NPAD2 + c] * w);
  }
}

__global__ void k_final(const float* __restrict__ agg2, const float* __restrict__ b2,
                        float* __restrict__ out) {
  int idx = blockIdx.x * blockDim.x + threadIdx.x;
  if (idx < N_NODES * C_OUT) {
    int c = idx % C_OUT;
    out[idx] = agg2[idx] + b2[c];
  }
}

// ---------------------------------------------------------------------------
// Host-side launcher
// ---------------------------------------------------------------------------
static inline size_t alignup(size_t x) { return (x + 255) & ~(size_t)255; }

extern "C" void kernel_launch(void* const* d_in, const int* in_sizes, int n_in,
                              void* d_out, int out_size, void* d_ws, size_t ws_size,
                              hipStream_t stream) {
  const float* x  = (const float*)d_in[0];                 // [50000,1433]
  const int*   ei = (const int*)d_in[1];                   // [2,1600000]
  const float* W1 = (const float*)d_in[2];                 // [1433,256]
  const float* b1 = (const float*)d_in[3];                 // [256]
  const float* W2 = (const float*)d_in[4];                 // [256,7]
  const float* b2 = (const float*)d_in[5];                 // [7]
  float* out = (float*)d_out;                              // [50000,7]

  const int* src = ei;
  const int* dst = ei + N_EDGES;

  // Workspace carve-up
  char* base = (char*)d_ws;
  size_t off = 0;
  float* deg  = (float*)(base + off); off += alignup((size_t)N_NODES * 4);
  float* dinv = (float*)(base + off); off += alignup((size_t)N_NODES * 4);
  float* h1   = (float*)(base + off); off += alignup((size_t)N_NODES * HID * 4);   // X@W1, later h2
  float* agg1 = (float*)(base + off); off += alignup((size_t)N_NODES * HID * 4);
  float* g    = (float*)(base + off); off += alignup((size_t)N_NODES * NPAD2 * 4); // h2@W2p
  float* agg2 = (float*)(base + off); off += alignup((size_t)N_NODES * C_OUT * 4);
  float* W2p  = (float*)(base + off); off += alignup((size_t)HID * NPAD2 * 4);
  (void)ws_size; (void)n_in; (void)in_sizes; (void)out_size;

  // --- degrees & norms ---
  k_init_deg <<<(N_NODES + 255) / 256, 256, 0, stream>>>(deg, N_NODES);
  k_deg_edges<<<(N_EDGES + 255) / 256, 256, 0, stream>>>(dst, deg, N_EDGES);
  k_dinv     <<<(N_NODES + 255) / 256, 256, 0, stream>>>(deg, dinv, N_NODES);

  // --- pad W2 ---
  k_pad_w2<<<(HID * NPAD2 + 255) / 256, 256, 0, stream>>>(W2, W2p);

  // --- layer 1: h1 = X @ W1 (WMMA) ---
  {
    dim3 grid((N_NODES + BM - 1) / BM, HID / BN);   // 391 x 4
    gcn_gemm_bf16x3<<<grid, 256, 0, stream>>>(x, F_IN, W1, HID, h1, HID,
                                              N_NODES, HID, F_IN);
  }
  // --- layer 1 aggregation + bias + relu ---
  k_selfloop1<<<N_NODES, 256, 0, stream>>>(h1, dinv, agg1);
  k_scatter1 <<<(N_EDGES + EDGES_PER_BLOCK - 1) / EDGES_PER_BLOCK, 256, 0, stream>>>(
      src, dst, dinv, h1, agg1, N_EDGES);
  k_bias_relu1<<<N_NODES, 256, 0, stream>>>(agg1, b1, h1);   // h1 now holds h2

  // --- layer 2: g = h2 @ W2p (WMMA) ---
  {
    dim3 grid((N_NODES + BM - 1) / BM, NPAD2 / BN); // 391 x 1
    gcn_gemm_bf16x3<<<grid, 256, 0, stream>>>(h1, HID, W2p, NPAD2, g, NPAD2,
                                              N_NODES, NPAD2, HID);
  }
  // --- layer 2 aggregation + bias ---
  k_selfloop2<<<(N_NODES * C_OUT + 255) / 256, 256, 0, stream>>>(g, dinv, agg2);
  k_scatter2 <<<(N_EDGES + 255) / 256, 256, 0, stream>>>(src, dst, dinv, g, agg2, N_EDGES);
  k_final    <<<(N_NODES * C_OUT + 255) / 256, 256, 0, stream>>>(agg2, b2, out);
}
